// CausalSelfAttention_12266426597861
// MI455X (gfx1250) — matrix operand order repaired and checked by reading, compiled, not run on hardware
//
#include <hip/hip_runtime.h>

// ---------------------------------------------------------------------------
// Causal self-attention for MI455X (gfx1250, wave32, WMMA bf16 16x16x32).
// B=2, T=4096, D=512, H=8, DH=64.
// Stage 0: convert x -> bf16; transpose+convert w_qkv, w_proj -> bf16
// Stage 1: qkv = x @ w_qkv + b_qkv   (Q,K row-major bf16; V stored transposed)
// Stage 2: flash attention, wave per 16-row Q tile; all fragment traffic is
//          b128; softmax row-sums computed on the matrix pipe (P @ ones).
// Stage 3: out = attn @ w_proj + b_proj (f32 out)
// ---------------------------------------------------------------------------

typedef __attribute__((ext_vector_type(16))) __bf16 v16bf;
typedef __attribute__((ext_vector_type(8)))  float  v8f;

constexpr int Bc  = 2;
constexpr int T_  = 4096;
constexpr int D_  = 512;
constexpr int Hh  = 8;
constexpr int DHd = 64;
constexpr int M_  = Bc * T_;     // 8192 rows
constexpr int N1  = 3 * D_;      // 1536 qkv cols
constexpr int SD  = N1;          // qkv row stride

__device__ __forceinline__ v8f wmma_bf16(v16bf a, v16bf b, v8f c) {
  return __builtin_amdgcn_wmma_f32_16x16x32_bf16(false, a, false, b, (short)0, c,
                                                 false, false);
}

// A-frag (16xK 16-bit): halves 0..15 -> K = base..base+15
// with base = 16*(lane>=16).  Two contiguous 16B runs, 8 halves apart.
__device__ __forceinline__ int kAbase(int lane) { return (lane >> 4) << 4; }
// B-frag (Kx16 16-bit): halves 0..15 -> K = base..base+15, base = 16*(lane>=16).
__device__ __forceinline__ int kBbase(int lane) { return (lane >> 4) << 4; }

// Build a v16bf fragment from two 16-byte chunks (each 8 bf16 halves).
__device__ __forceinline__ v16bf load_frag2(const __bf16* p0, const __bf16* p1) {
  union { uint4 q[2]; v16bf v; } u;
  u.q[0] = *(const uint4*)p0;
  u.q[1] = *(const uint4*)p1;
  return u.v;
}

// ---------------------------------------------------------------------------
// Stage 0 prep kernels
// ---------------------------------------------------------------------------
__global__ void __launch_bounds__(256)
cvt_bf16_kernel(const float* __restrict__ in, __bf16* __restrict__ out, int n) {
  const int stride = gridDim.x * blockDim.x;
  for (int i = blockIdx.x * blockDim.x + threadIdx.x; i < n; i += stride)
    out[i] = (__bf16)in[i];
}

// in[K][N] (f32) -> out[N][K] (bf16); reads coalesced over n.
__global__ void __launch_bounds__(256)
transpose_bf16_kernel(const float* __restrict__ in, __bf16* __restrict__ out,
                      int K, int N) {
  const int total = K * N;
  const int stride = gridDim.x * blockDim.x;
  for (int idx = blockIdx.x * blockDim.x + threadIdx.x; idx < total; idx += stride) {
    const int k = idx / N;
    const int n = idx - k * N;
    out[(size_t)n * K + k] = (__bf16)in[idx];
  }
}

// ---------------------------------------------------------------------------
// Stage 1: QKV GEMM.  A = xbf (M x 512 bf16), B = WtQKV (1536 x 512 bf16,
// n-major so K is contiguous).  Q,K cols -> qkv buffer; V cols -> Vt transposed.
// One wave per 16x16 tile; per K-step: 4 x global_load_b128 + 1 WMMA.
// ---------------------------------------------------------------------------
__global__ void __launch_bounds__(128)
qkv_gemm_kernel(const __bf16* __restrict__ A, const __bf16* __restrict__ Wt,
                const float* __restrict__ bias, __bf16* __restrict__ Out,
                __bf16* __restrict__ Vt) {
  const int lane = threadIdx.x & 31;
  const int wave = blockIdx.x * (blockDim.x >> 5) + (threadIdx.x >> 5);
  constexpr int nT = N1 >> 4;                 // 96
  const int mT = wave / nT;
  const int nI = wave - mT * nT;
  if (mT >= (M_ >> 4)) return;                // wave-uniform
  const int m0 = mT << 4, n0 = nI << 4;
  const int l16  = lane & 15;
  const int rowA = m0 + l16;
  const int colB = n0 + l16;
  const int ka = kAbase(lane), kb = kBbase(lane);

  const __bf16* ap = A + (size_t)rowA * D_ + ka;
  const __bf16* bp = Wt + (size_t)colB * D_ + kb;

  v8f acc = {};
#pragma unroll 4
  for (int k0 = 0; k0 < D_; k0 += 32) {
    const v16bf a = load_frag2(ap + k0, ap + k0 + 8);
    const v16bf b = load_frag2(bp + k0, bp + k0 + 8);
    acc = wmma_bf16(a, b, acc);
  }
  const float bv = bias[colB];
  if (n0 < 2 * D_) {                          // Q,K region: row-major store
#pragma unroll
    for (int r = 0; r < 8; ++r) {
      const int row = m0 + r + ((lane >> 4) << 3);
      Out[(size_t)row * SD + colB] = (__bf16)(acc[r] + bv);
    }
  } else {                                    // V region: store transposed
    const int vcol = colB - 2 * D_;           // h*64 + d in [0,512)
#pragma unroll
    for (int r = 0; r < 8; ++r) {
      const int row = m0 + r + ((lane >> 4) << 3);
      const int b_ = row >> 12;               // / T_
      const int t  = row & (T_ - 1);
      Vt[(size_t)(b_ * D_ + vcol) * T_ + t] = (__bf16)(acc[r] + bv);
    }
  }
}

// ---------------------------------------------------------------------------
// Stage 2: flash attention.  Wave per 16-row Q tile per (b,h).
// ---------------------------------------------------------------------------
__global__ void __launch_bounds__(128)
attn_kernel(const __bf16* __restrict__ qkv, const __bf16* __restrict__ Vt,
            __bf16* __restrict__ outA) {
  __shared__ __align__(16) __bf16 pbuf[4][16 * 32];   // per-wave P staging
  const int lane = threadIdx.x & 31;
  const int wv   = threadIdx.x >> 5;
  const int tile = blockIdx.x * (blockDim.x >> 5) + wv;
  constexpr int qTiles = T_ >> 4;             // 256
  const int qi = tile % qTiles;
  const int hd = (tile / qTiles) % Hh;
  const int b_ = tile / (qTiles * Hh);
  if (b_ >= Bc) return;                       // wave-uniform

  const int q0  = qi << 4;
  const int l16 = lane & 15;
  const int ka = kAbase(lane), kb = kBbase(lane);
  const float scale = 0.125f;                 // 1/sqrt(64)
  const size_t rowBase = (size_t)b_ * T_;

  const __bf16* Qb = qkv + 0 * D_ + hd * DHd;
  const __bf16* Kb = qkv + 1 * D_ + hd * DHd;
  const __bf16* Vtb = Vt + (size_t)(b_ * D_ + hd * DHd) * T_;

  // Q tile 16x64 as two A-fragments (2 x b128 each), hoisted.
  v16bf qf[2];
#pragma unroll
  for (int s = 0; s < 2; ++s) {
    const __bf16* p = Qb + (rowBase + q0 + l16) * SD + s * 32 + ka;
    qf[s] = load_frag2(p, p + 8);
  }

  // All-ones B-fragment: P(16x32) @ ones(32x16) puts each row-sum of P in the
  // same C-layout slot as lrow[r] -- row reduction on the matrix pipe.
  v16bf ones;
#pragma unroll
  for (int h = 0; h < 16; ++h) ones[h] = (__bf16)1.0f;

  v8f o[4] = {};
  float mrow[8], lrow[8];
#pragma unroll
  for (int r = 0; r < 8; ++r) { mrow[r] = -3.0e38f; lrow[r] = 0.0f; }

  const int nsteps = (q0 + 16 + 31) >> 5;
  for (int j = 0; j < nsteps; ++j) {
    const int kb0 = j << 5;
    const __bf16* kp0 = Kb + (rowBase + kb0 + l16) * SD + kb;
    const __bf16* kp1 = Kb + (rowBase + kb0 + 16 + l16) * SD + kb;

    // Prefetch next step's K/V rows into cache (global_prefetch_b8).
    __builtin_prefetch(kp0 + 32 * SD, 0, 1);
    __builtin_prefetch(kp1 + 32 * SD, 0, 1);
    __builtin_prefetch(Vtb + (size_t)l16 * T_ + kb0 + 32, 0, 1);

    // ---- S = Q @ K^T : 2 n-tiles x 2 K-steps = 4 WMMAs, 8 b128 loads ----
    v8f s0 = {}, s1 = {};
#pragma unroll
    for (int ss = 0; ss < 2; ++ss) {
      const v16bf kf0 = load_frag2(kp0 + ss * 32, kp0 + ss * 32 + 8);
      const v16bf kf1 = load_frag2(kp1 + ss * 32, kp1 + ss * 32 + 8);
      s0 = wmma_bf16(qf[ss], kf0, s0);
      s1 = wmma_bf16(qf[ss], kf1, s1);
    }

    // ---- causal mask + online max (shuffle reduce over the 16-lane half) ----
    float al[8];
#pragma unroll
    for (int r = 0; r < 8; ++r) {
      const int qg = q0 + r + ((lane >> 4) << 3);
      float v0 = s0[r] * scale;
      float v1 = s1[r] * scale;
      if (kb0 + l16 > qg)      v0 = -3.0e38f;
      if (kb0 + 16 + l16 > qg) v1 = -3.0e38f;
      float mx = fmaxf(v0, v1);
      mx = fmaxf(mx, __shfl_xor(mx, 1, 32));
      mx = fmaxf(mx, __shfl_xor(mx, 2, 32));
      mx = fmaxf(mx, __shfl_xor(mx, 4, 32));
      mx = fmaxf(mx, __shfl_xor(mx, 8, 32));
      const float nm = fmaxf(mrow[r], mx);
      al[r] = __expf(mrow[r] - nm);
      s0[r] = __expf(v0 - nm);
      s1[r] = __expf(v1 - nm);
      mrow[r] = nm;
    }
#pragma unroll
    for (int t = 0; t < 4; ++t)
#pragma unroll
      for (int r = 0; r < 8; ++r) o[t][r] *= al[r];

    // ---- P: C-layout -> A-layout via per-wave LDS ----
#pragma unroll
    for (int r = 0; r < 8; ++r) {
      const int row = r + ((lane >> 4) << 3);
      pbuf[wv][row * 32 + l16]      = (__bf16)s0[r];
      pbuf[wv][row * 32 + 16 + l16] = (__bf16)s1[r];
    }
    asm volatile("s_wait_dscnt 0" ::: "memory");
    const v16bf pf =
        load_frag2(&pbuf[wv][l16 * 32 + ka], &pbuf[wv][l16 * 32 + ka + 8]);
    asm volatile("" ::: "memory");

    // ---- row sums of (bf16-rounded) P on the matrix pipe ----
    v8f srow = {};
    srow = wmma_bf16(pf, ones, srow);
#pragma unroll
    for (int r = 0; r < 8; ++r) lrow[r] = lrow[r] * al[r] + srow[r];

    // ---- O += P @ V : 4 WMMAs, 8 b128 loads from transposed V ----
#pragma unroll
    for (int t = 0; t < 4; ++t) {
      const __bf16* vp = Vtb + (size_t)(t * 16 + l16) * T_ + kb0 + kb;
      const v16bf vf = load_frag2(vp, vp + 8);
      o[t] = wmma_bf16(pf, vf, o[t]);
    }
  }

  // ---- finalize ----
#pragma unroll
  for (int r = 0; r < 8; ++r) {
    const float inv = 1.0f / lrow[r];
    const int qg = q0 + r + ((lane >> 4) << 3);
#pragma unroll
    for (int t = 0; t < 4; ++t)
      outA[(rowBase + qg) * (size_t)D_ + hd * DHd + t * 16 + l16] =
          (__bf16)(o[t][r] * inv);
  }
}

// ---------------------------------------------------------------------------
// Stage 3: output projection.  A = attn (bf16), B = WtProj (512x512 bf16).
// ---------------------------------------------------------------------------
__global__ void __launch_bounds__(128)
proj_gemm_kernel(const __bf16* __restrict__ A, const __bf16* __restrict__ Wt,
                 const float* __restrict__ bias, float* __restrict__ Out) {
  const int lane = threadIdx.x & 31;
  const int wave = blockIdx.x * (blockDim.x >> 5) + (threadIdx.x >> 5);
  constexpr int nT = D_ >> 4;                 // 32
  const int mT = wave / nT;
  const int nI = wave - mT * nT;
  if (mT >= (M_ >> 4)) return;
  const int m0 = mT << 4, n0 = nI << 4;
  const int l16  = lane & 15;
  const int rowA = m0 + l16;
  const int colB = n0 + l16;
  const int ka = kAbase(lane), kb = kBbase(lane);

  const __bf16* ap = A + (size_t)rowA * D_ + ka;
  const __bf16* bp = Wt + (size_t)colB * D_ + kb;

  v8f acc = {};
#pragma unroll 4
  for (int k0 = 0; k0 < D_; k0 += 32) {
    const v16bf a = load_frag2(ap + k0, ap + k0 + 8);
    const v16bf b = load_frag2(bp + k0, bp + k0 + 8);
    acc = wmma_bf16(a, b, acc);
  }
  const float bv = bias[colB];
#pragma unroll
  for (int r = 0; r < 8; ++r) {
    const int row = m0 + r + ((lane >> 4) << 3);
    Out[(size_t)row * D_ + colB] = acc[r] + bv;
  }
}

// ---------------------------------------------------------------------------
extern "C" void kernel_launch(void* const* d_in, const int* in_sizes, int n_in,
                              void* d_out, int out_size, void* d_ws, size_t ws_size,
                              hipStream_t stream) {
  (void)in_sizes; (void)n_in; (void)out_size; (void)ws_size;

  const float* x      = (const float*)d_in[0];
  const float* w_qkv  = (const float*)d_in[1];
  const float* b_qkv  = (const float*)d_in[2];
  const float* w_proj = (const float*)d_in[3];
  const float* b_proj = (const float*)d_in[4];
  float* out = (float*)d_out;

  // Workspace layout (bf16 elements), all 16B aligned.
  __bf16* qkvbuf  = (__bf16*)d_ws;                        // M_*N1   (Q,K used)
  __bf16* attnbuf = qkvbuf  + (size_t)M_ * N1;            // M_*D_
  __bf16* vtbuf   = attnbuf + (size_t)M_ * D_;            // Bc*D_*T_
  __bf16* xbf     = vtbuf   + (size_t)Bc * D_ * T_;       // M_*D_
  __bf16* wtq     = xbf     + (size_t)M_ * D_;            // N1*D_
  __bf16* wtp     = wtq     + (size_t)N1 * D_;            // D_*D_

  // Stage 0: convert / transpose (tiny vs. attention compute).
  cvt_bf16_kernel<<<2048, 256, 0, stream>>>(x, xbf, M_ * D_);
  transpose_bf16_kernel<<<1536, 256, 0, stream>>>(w_qkv, wtq, D_, N1);
  transpose_bf16_kernel<<<512, 256, 0, stream>>>(w_proj, wtp, D_, D_);

  // Stage 1: 512*96 tile-waves, 4 waves/block.
  qkv_gemm_kernel<<<(M_ / 16) * (N1 / 16) / 4, 128, 0, stream>>>(
      xbf, wtq, b_qkv, qkvbuf, vtbuf);

  // Stage 2: B*H*(T/16) = 4096 waves.
  attn_kernel<<<Bc * Hh * (T_ / 16) / 4, 128, 0, stream>>>(qkvbuf, vtbuf, attnbuf);

  // Stage 3: 512*32 tile-waves.
  proj_gemm_kernel<<<(M_ / 16) * (D_ / 16) / 4, 128, 0, stream>>>(
      attnbuf, wtp, b_proj, out);
}
